// ResidualVQ_10479720202873
// MI455X (gfx1250) — matrix operand order repaired and checked by reading, not compiled, not run on hardware
//
#include <hip/hip_runtime.h>
#include <hip/hip_bf16.h>
#include <math.h>

// Problem constants (from reference): x[32,512,2048], codebooks[6,512,512]
#define NB     32
#define CD     512      // C == code_dim == nb_code
#define TD     2048
#define QL     6
#define NT     65536    // NB*TD rows
#define OUTQ   33554432 // NT*CD quantized_out elements
#define OUTI   393216   // NT*QL index elements

#define LDS_STRIDE 520  // 512 + 8 bf16 pad -> 4-dword bank skew per row

#if defined(__has_builtin)
#if __has_builtin(__builtin_amdgcn_global_load_async_to_lds_b128) && \
    __has_builtin(__builtin_amdgcn_s_wait_asynccnt)
#define USE_ASYNC_LDS 1
#endif
#endif

typedef __attribute__((ext_vector_type(4)))  unsigned int v4u;
typedef __attribute__((ext_vector_type(4)))  int          v4i;
typedef __attribute__((ext_vector_type(16))) __bf16       v16bf;
typedef __attribute__((ext_vector_type(8)))  float        v8f;

typedef __attribute__((address_space(1))) v4i* gptr_v4i;   // global
typedef __attribute__((address_space(3))) v4i* lptr_v4i;   // LDS

union BFrag { v4u u[2]; v16bf v; };

__device__ __forceinline__ unsigned short f2bf(float f) {
    unsigned int u = __float_as_uint(f);
    unsigned int r = u + 0x7FFFu + ((u >> 16) & 1u);   // round-to-nearest-even
    return (unsigned short)(r >> 16);
}

// ---------------------------------------------------------------------------
// 1) Codebooks: f32 -> bf16 copy + per-code squared norm. One block per code row.
// ---------------------------------------------------------------------------
__global__ __launch_bounds__(256)
void prep_codebook_kernel(const float* __restrict__ cb,
                          unsigned short* __restrict__ cbb,
                          float* __restrict__ cbn) {
    const int row = blockIdx.x;                  // 0 .. QL*CD-1
    const float* src = cb + (size_t)row * CD;
    unsigned short* dst = cbb + (size_t)row * CD;
    const int t = threadIdx.x;
    float a = src[t], b = src[t + 256];
    dst[t] = f2bf(a); dst[t + 256] = f2bf(b);
    __shared__ float red[256];
    red[t] = a * a + b * b;
    __syncthreads();
    for (int s = 128; s > 0; s >>= 1) { if (t < s) red[t] += red[t + s]; __syncthreads(); }
    if (t == 0) cbn[row] = red[0];
}

// ---------------------------------------------------------------------------
// 2) Transpose x[N,C,T] -> residual[NT,C] (f32 + bf16 shadow), LDS tiled.
// ---------------------------------------------------------------------------
__global__ __launch_bounds__(256)
void init_residual_kernel(const float* __restrict__ x,
                          float* __restrict__ rf,
                          unsigned short* __restrict__ rb) {
    __shared__ float tile[32][33];
    const int n = blockIdx.z, tB = blockIdx.x * 32, cB = blockIdx.y * 32;
    const int tx = threadIdx.x, ty = threadIdx.y;
    const float* xp = x + ((size_t)n * CD + cB) * TD + tB;
#pragma unroll
    for (int i = 0; i < 32; i += 8)
        tile[ty + i][tx] = xp[(size_t)(ty + i) * TD + tx];      // coalesced along t
    __syncthreads();
#pragma unroll
    for (int i = 0; i < 32; i += 8) {
        const int t = tB + ty + i;
        const size_t o = ((size_t)n * TD + t) * CD + cB + tx;   // coalesced along c
        const float v = tile[tx][ty + i];
        rf[o] = v; rb[o] = f2bf(v);
    }
}

// ---------------------------------------------------------------------------
// 3) Zero per-run accumulators (counts histogram + 3 scalars).
// ---------------------------------------------------------------------------
__global__ void zero_misc_kernel(unsigned int* counts, float* scal) {
    counts[threadIdx.x] = 0u;
    if (threadIdx.x < 3) scal[threadIdx.x] = 0.0f;
}

// ---------------------------------------------------------------------------
// 4) Argmin over K=512 codes via V_WMMA_F32_16X16X32_BF16.
//    Block = 8 waves x 16 rows = 128 rows. B tile (16 codes x 512 dims bf16,
//    16KB) staged in LDS once per block and shared by all 8 waves (8x VMEM
//    reduction), double-buffered with async global->LDS loads (ASYNCcnt).
//    score(row,code) = ||cb||^2 - 2 * <r, cb>   (||r||^2 constant per row)
// ---------------------------------------------------------------------------
__global__ __launch_bounds__(256)
void argmin_kernel(const unsigned short* __restrict__ rb,
                   const unsigned short* __restrict__ cbb,   // this layer, bf16
                   const float* __restrict__ cbn,            // this layer norms
                   unsigned int* __restrict__ idxOut) {
    __shared__ unsigned short btile[2][16 * LDS_STRIDE];

    const int tid = threadIdx.x;
    const int wave = tid >> 5, lane = tid & 31;
    const int rowBase = blockIdx.x * 128 + wave * 16;
    const int m = lane & 15, half = lane >> 4;

    // Preload A: 16 rows x K=512 bf16 as 16 fragments (ISA 16-bit A layout).
    BFrag a[16];
    const unsigned short* rrow = rb + (size_t)(rowBase + m) * CD;
#pragma unroll
    for (int kc = 0; kc < 16; ++kc) {
        const int off = kc * 32 + 8 * half;
        a[kc].u[0] = *reinterpret_cast<const v4u*>(rrow + off);        // K off..off+7
        a[kc].u[1] = *reinterpret_cast<const v4u*>(rrow + off + 16);   // K off+16..+23
    }

#if USE_ASYNC_LDS
    // Cooperative async fill of one 16x512 bf16 B tile (1024 x b128 chunks).
    auto issue_tile = [&](int nb, int buf) {
#pragma unroll
        for (int p = 0; p < 4; ++p) {
            const int ch = tid + p * 256;            // 0..1023
            const int r  = ch >> 6;                  // code row 0..15
            const int c8 = (ch & 63) << 3;           // elem col, 8 bf16 per chunk
            const unsigned short* g = cbb + (size_t)(nb * 16 + r) * CD + c8;
            unsigned short* l = &btile[buf][r * LDS_STRIDE + c8];
            __builtin_amdgcn_global_load_async_to_lds_b128(
                (gptr_v4i)(void*)const_cast<unsigned short*>(g),
                (lptr_v4i)(void*)l, 0, 0);
        }
    };
    issue_tile(0, 0);
    __builtin_amdgcn_s_wait_asynccnt(0);
    __syncthreads();
#else
    // Fallback: stage through registers, ds_store_b128 into LDS.
    {
#pragma unroll
        for (int p = 0; p < 4; ++p) {
            const int ch = tid + p * 256;
            const int r  = ch >> 6;
            const int c8 = (ch & 63) << 3;
            v4u v = *reinterpret_cast<const v4u*>(cbb + (size_t)r * CD + c8);
            *reinterpret_cast<v4u*>(&btile[0][r * LDS_STRIDE + c8]) = v;
        }
    }
    __syncthreads();
#endif

    float bestv[8];
    int   besti[8];
#pragma unroll
    for (int i = 0; i < 8; ++i) { bestv[i] = 3.4e38f; besti[i] = 0; }

    for (int nb = 0; nb < 32; ++nb) {            // 32 code tiles of 16
        const int buf = nb & 1;
#if USE_ASYNC_LDS
        if (nb + 1 < 32) issue_tile(nb + 1, buf ^ 1);   // overlap with compute
#else
        v4u st[4];
        if (nb + 1 < 32) {
#pragma unroll
            for (int p = 0; p < 4; ++p) {
                const int ch = tid + p * 256;
                const int r  = ch >> 6;
                const int c8 = (ch & 63) << 3;
                st[p] = *reinterpret_cast<const v4u*>(
                    cbb + (size_t)((nb + 1) * 16 + r) * CD + c8);
            }
        }
#endif
        if (nb + 2 < 32)  // keep next-next tile flowing toward L2/L0
            __builtin_prefetch(cbb + (size_t)((nb + 2) * 16 + (lane & 15)) * CD, 0, 0);

        const int code = nb * 16 + (lane & 15);
        const unsigned short* lrow = &btile[buf][(lane & 15) * LDS_STRIDE];
        v8f acc = {0.f, 0.f, 0.f, 0.f, 0.f, 0.f, 0.f, 0.f};
#pragma unroll
        for (int kc = 0; kc < 16; ++kc) {        // K=512 in steps of 32
            BFrag b;
            const int off = kc * 32 + 8 * half;  // ISA B layout: lanes 16-31 hold K+8 / K+24
            b.u[0] = *reinterpret_cast<const v4u*>(lrow + off);
            b.u[1] = *reinterpret_cast<const v4u*>(lrow + off + 16);
            acc = __builtin_amdgcn_wmma_f32_16x16x32_bf16(
                false, a[kc].v, false, b.v, (short)0, acc, false, false);
        }
        const float cn = cbn[code];
#pragma unroll
        for (int i = 0; i < 8; ++i) {            // element (m=i+8*half, n=lane&15)
            const float s = cn - 2.0f * acc[i];
            if (s < bestv[i]) { bestv[i] = s; besti[i] = code; }   // first-min wins
        }

#if USE_ASYNC_LDS
        __builtin_amdgcn_s_wait_asynccnt(0);
#else
        if (nb + 1 < 32) {
#pragma unroll
            for (int p = 0; p < 4; ++p) {
                const int ch = tid + p * 256;
                const int r  = ch >> 6;
                const int c8 = (ch & 63) << 3;
                *reinterpret_cast<v4u*>(&btile[buf ^ 1][r * LDS_STRIDE + c8]) = st[p];
            }
        }
#endif
        __syncthreads();
    }

    // Reduce the 16 n-lanes of each half (xor masks < 16 stay inside the half).
#pragma unroll
    for (int d = 1; d < 16; d <<= 1) {
#pragma unroll
        for (int i = 0; i < 8; ++i) {
            const float ov = __shfl_xor(bestv[i], d, 32);
            const int   oi = __shfl_xor(besti[i], d, 32);
            if (ov < bestv[i] || (ov == bestv[i] && oi < besti[i])) {
                bestv[i] = ov; besti[i] = oi;
            }
        }
    }
    if ((lane & 15) == 0) {
#pragma unroll
        for (int i = 0; i < 8; ++i)
            idxOut[rowBase + i + 8 * half] = (unsigned int)besti[i];
    }
}

// ---------------------------------------------------------------------------
// 5) Residual update: gather f32 code row, r -= cb, refresh bf16 shadow,
//    commit += ||r_new||^2, histogram, write float index output.
// ---------------------------------------------------------------------------
__global__ __launch_bounds__(256)
void update_kernel(float* __restrict__ rf,
                   unsigned short* __restrict__ rb,
                   const float* __restrict__ cb,       // this layer, f32
                   const unsigned int* __restrict__ idx,
                   unsigned int* __restrict__ counts,
                   float* __restrict__ scal,           // scal[0] = commit acc
                   float* __restrict__ idxOutF, int q) {
    const int row = blockIdx.x, t = threadIdx.x;
    const unsigned int code = idx[row];
    const float* crow = cb + (size_t)code * CD;
    const size_t base = (size_t)row * CD;
    float local = 0.0f;
#pragma unroll
    for (int e = t; e < CD; e += 256) {
        const float rv = rf[base + e] - crow[e];
        rf[base + e] = rv;
        rb[base + e] = f2bf(rv);
        local += rv * rv;
    }
    __shared__ float red[256];
    red[t] = local; __syncthreads();
    for (int s = 128; s > 0; s >>= 1) { if (t < s) red[t] += red[t + s]; __syncthreads(); }
    if (t == 0) {
        atomicAdd(&scal[0], red[0]);
        atomicAdd(&counts[code], 1u);
        idxOutF[(size_t)row * QL + q] = (float)code;
    }
}

// ---------------------------------------------------------------------------
// 6) Perplexity + loss accumulation for one layer; single block of 512.
// ---------------------------------------------------------------------------
__global__ __launch_bounds__(512)
void perp_kernel(unsigned int* __restrict__ counts,
                 float* __restrict__ scal,   // [0]=commit, [1]=lossAcc, [2]=perpAcc
                 int q, float* __restrict__ outTail) {
    const int t = threadIdx.x;
    const float prob = (float)counts[t] * (1.0f / (float)NT);
    __shared__ float red[512];
    red[t] = prob * logf(prob + 1e-7f);
    __syncthreads();
    for (int s = 256; s > 0; s >>= 1) { if (t < s) red[t] += red[t + s]; __syncthreads(); }
    counts[t] = 0u;                                  // reset for next layer
    if (t == 0) {
        scal[2] += expf(-red[0]);                    // perplexity
        scal[1] += scal[0] * (1.0f / (float)OUTQ);   // mse over N*T*C elements
        scal[0] = 0.0f;
        if (q == QL - 1) {
            outTail[0] = scal[1] * (1.0f / (float)QL);  // all_losses
            outTail[1] = scal[2] * (1.0f / (float)QL);  // all_perplexity
        }
    }
}

// ---------------------------------------------------------------------------
// 7) quantized_out = x - residual_final, written back in [N,C,T] layout.
// ---------------------------------------------------------------------------
__global__ __launch_bounds__(256)
void finalize_kernel(const float* __restrict__ x,
                     const float* __restrict__ rf,
                     float* __restrict__ out) {
    __shared__ float tile[32][33];
    const int n = blockIdx.z, tB = blockIdx.x * 32, cB = blockIdx.y * 32;
    const int tx = threadIdx.x, ty = threadIdx.y;
#pragma unroll
    for (int i = 0; i < 32; i += 8) {
        const int t = tB + ty + i;
        tile[ty + i][tx] = rf[((size_t)n * TD + t) * CD + cB + tx];  // coalesced on c
    }
    __syncthreads();
    const size_t xo = ((size_t)n * CD + cB) * TD + tB;
#pragma unroll
    for (int i = 0; i < 32; i += 8) {
        const size_t o = xo + (size_t)(ty + i) * TD + tx;            // coalesced on t
        out[o] = x[o] - tile[tx][ty + i];
    }
}

// ---------------------------------------------------------------------------
extern "C" void kernel_launch(void* const* d_in, const int* in_sizes, int n_in,
                              void* d_out, int out_size, void* d_ws, size_t ws_size,
                              hipStream_t stream) {
    const float* x  = (const float*)d_in[0];   // [32,512,2048]
    const float* cb = (const float*)d_in[1];   // [6,512,512]
    float* out = (float*)d_out;
    float* outIdx  = out + OUTQ;
    float* outTail = out + OUTQ + OUTI;

    // Workspace carve-up (~205 MB)
    char* w = (char*)d_ws;
    float*          rf     = (float*)w;          w += (size_t)NT * CD * 4;      // 134 MB
    unsigned short* rb     = (unsigned short*)w; w += (size_t)NT * CD * 2;      //  67 MB
    unsigned short* cbb    = (unsigned short*)w; w += (size_t)QL * CD * CD * 2; //   3 MB
    float*          cbn    = (float*)w;          w += (size_t)QL * CD * 4;
    unsigned int*   idx    = (unsigned int*)w;   w += (size_t)NT * 4;
    unsigned int*   counts = (unsigned int*)w;   w += 512 * 4;
    float*          scal   = (float*)w;          w += 64;

    prep_codebook_kernel<<<QL * CD, 256, 0, stream>>>(cb, cbb, cbn);
    init_residual_kernel<<<dim3(TD / 32, CD / 32, NB), dim3(32, 8), 0, stream>>>(x, rf, rb);
    zero_misc_kernel<<<1, 512, 0, stream>>>(counts, scal);

    for (int q = 0; q < QL; ++q) {
        argmin_kernel<<<NT / 128, 256, 0, stream>>>(
            rb, cbb + (size_t)q * CD * CD, cbn + (size_t)q * CD, idx);
        update_kernel<<<NT, 256, 0, stream>>>(
            rf, rb, cb + (size_t)q * CD * CD, idx, counts, scal, outIdx, q);
        perp_kernel<<<1, 512, 0, stream>>>(counts, scal, q, outTail);
    }

    finalize_kernel<<<dim3(TD / 32, CD / 32, NB), dim3(32, 8), 0, stream>>>(x, rf, out);
}
